// ClassAEBottleneck_6734508720575
// MI455X (gfx1250) — matrix operand order repaired and checked by reading, not compiled, not run on hardware
//
#include <hip/hip_runtime.h>

// Problem dims (from reference)
#define B_SZ  8192
#define DIN   1024
#define DH    2048
#define DBOT  512
#define DEXP  1024
#define NCLS  10

typedef __attribute__((ext_vector_type(16))) __bf16 v16bf;
typedef __attribute__((ext_vector_type(2)))  __bf16 v2bf;
typedef __attribute__((ext_vector_type(8)))  float  v8f;
typedef __attribute__((ext_vector_type(2)))  float  f32x2;
typedef __attribute__((ext_vector_type(4)))  unsigned int u32x4;

// Packed hardware conversion: float pair -> packed bf16 pair (v_cvt_pk_bf16_f32),
// two pairs written as a single 64-bit LDS store. No scalar repack movs.
static __device__ __forceinline__ void store_bf16x4(__bf16* dst,
                                                    float a, float b, float c, float d) {
  union { v2bf h[2]; unsigned long long u; } p;
  p.h[0] = __builtin_convertvector((f32x2){a, b}, v2bf);
  p.h[1] = __builtin_convertvector((f32x2){c, d}, v2bf);
  *(unsigned long long*)dst = p.u;
}

// ---------------- Routing kernels (label -> grouped row permutation) ----------------
__global__ void route_init(int* __restrict__ cnt, int* __restrict__ cur) {
  int t = threadIdx.x;
  if (t < NCLS) { cnt[t] = 0; cur[t] = 0; }
}
__global__ void route_hist(const int* __restrict__ labels, int* __restrict__ cnt, int n) {
  int i = blockIdx.x * blockDim.x + threadIdx.x;
  if (i < n) atomicAdd(&cnt[labels[i]], 1);
}
__global__ void route_scan(const int* __restrict__ cnt, int* __restrict__ off) {
  if (threadIdx.x == 0) {
    int s = 0;
    for (int e = 0; e < NCLS; ++e) { off[e] = s; s += cnt[e]; }
  }
}
__global__ void route_scatter(const int* __restrict__ labels, const int* __restrict__ off,
                              int* __restrict__ cur, int* __restrict__ perm, int n) {
  int i = blockIdx.x * blockDim.x + threadIdx.x;
  if (i < n) {
    int l = labels[i];
    int p = atomicAdd(&cur[l], 1);
    perm[off[l] + p] = i;
  }
}

// ---------------- Tiled bf16-WMMA GEMM:  C = act(A @ W + bias) ----------------
// A: [M,K] fp32 row-major, W: [K,N] fp32 row-major (per-expert offset when GATHER),
// C: [M,N] fp32. Block tile 128x128, K-step 32, double-buffered LDS, software
// pipelined: next tile's global loads issue before the WMMA block, converts and
// LDS stores land after it, one barrier per K-step. 256 threads = 8 waves (4Mx2N),
// each wave computes a 32x64 patch via 2x4 v_wmma_f32_16x16x32_bf16 tiles.
#define BM 128
#define BN 128
#define BK 32
#define KPAD 40   // LDS row stride in bf16 elements; multiple of 8 keeps 16B alignment
#define AI  ((BM * BK) / (256 * 4))   // 4 float4 chunks / thread for A tile
#define BI  ((BK * BN) / (256 * 4))   // 4 column chunks / thread for B tile

template<bool RELU, bool GATHER>
__global__ __launch_bounds__(256)
void gemm_bf16_wmma(const float* __restrict__ A, const float* __restrict__ W,
                    const float* __restrict__ bias, float* __restrict__ C,
                    int M, int N, int K,
                    const int* __restrict__ perm,
                    const int* __restrict__ cls_off,
                    const int* __restrict__ cls_cnt)
{
  __shared__ __bf16 As[2][BM * KPAD];   // A tile, bf16, row-major [m][k]
  __shared__ __bf16 Bs[2][BN * KPAD];   // W tile, bf16, TRANSPOSED [n][k]
  __shared__ int rmap[BM];

  const int tid    = threadIdx.x;
  const int lane   = tid & 31;        // wave32
  const int wave   = tid >> 5;
  const int waveM  = wave >> 1;       // 0..3 -> 32 rows each
  const int waveN  = wave & 1;        // 0..1 -> 64 cols each
  const int laneLo = lane & 15;
  const int laneHi = lane >> 4;

  const int m0 = blockIdx.x * BM;
  const int n0 = blockIdx.y * BN;

  if (GATHER) {
    const int e   = blockIdx.z;
    const int cnt = cls_cnt[e];
    if (m0 >= cnt) return;                       // uniform early-exit (before barriers)
    const int base = cls_off[e];
    W    += (size_t)e * K * N;
    bias += (size_t)e * N;
    if (tid < BM) {
      int r = m0 + tid;
      rmap[tid] = (r < cnt) ? perm[base + r] : -1;
    }
    __syncthreads();
  }

  // Per-thread staging coordinates (constant across K-steps)
  int arow[AI], acol[AI], agrow[AI];
  int bn[BI], bkq[BI];
  #pragma unroll
  for (int i = 0; i < AI; ++i) {
    int c = tid + 256 * i;
    arow[i] = c >> 3;                  // BK/4 = 8 float4 chunks per row
    acol[i] = (c & 7) << 2;
    agrow[i] = GATHER ? rmap[arow[i]] : (m0 + arow[i]);
  }
  #pragma unroll
  for (int i = 0; i < BI; ++i) {
    int c = tid + 256 * i;
    bn[i]  = c & (BN - 1);             // 0..127
    bkq[i] = (c >> 7) << 2;            // 0,4,...,28
  }

  float4 ga[AI];
  float  gb[BI][4];
  const int nk = K / BK;

  // ---- fetch tile kb into registers (global loads only, no waits forced) ----
  auto fetch = [&](int kb) {
    #pragma unroll
    for (int i = 0; i < AI; ++i) {
      ga[i] = make_float4(0.f, 0.f, 0.f, 0.f);
      if (agrow[i] >= 0)
        ga[i] = *(const float4*)(A + (size_t)agrow[i] * K + kb + acol[i]);
    }
    #pragma unroll
    for (int i = 0; i < BI; ++i) {
      const float* wp = W + (size_t)(kb + bkq[i]) * N + n0 + bn[i];
      gb[i][0] = wp[0];
      gb[i][1] = wp[(size_t)N];
      gb[i][2] = wp[(size_t)2 * N];
      gb[i][3] = wp[(size_t)3 * N];
    }
  };
  // ---- convert staged registers and store into LDS buffer p ----
  auto commit = [&](int p) {
    #pragma unroll
    for (int i = 0; i < AI; ++i)
      store_bf16x4(&As[p][arow[i] * KPAD + acol[i]], ga[i].x, ga[i].y, ga[i].z, ga[i].w);
    #pragma unroll
    for (int i = 0; i < BI; ++i)
      store_bf16x4(&Bs[p][bn[i] * KPAD + bkq[i]], gb[i][0], gb[i][1], gb[i][2], gb[i][3]);
  };

  v8f acc[2][4] = {};   // fp32 accumulators (8 VGPRs per 16x16 tile)
  union Frag { v16bf v; u32x4 u[2]; };
  Frag a[2], b[4];

  // Prologue: stage tile 0 into buffer 0
  fetch(0);
  commit(0);
  __syncthreads();

  for (int kt = 0; kt < nk; ++kt) {
    const int p = kt & 1;

    // Issue next tile's global loads now; the loadcnt wait sinks below the WMMAs.
    if (kt + 1 < nk) fetch((kt + 1) * BK);
    // L2 prefetch two tiles ahead (global_prefetch_b8)
    if (kt + 2 < nk)
      __builtin_prefetch(W + (size_t)((kt + 2) * BK + (tid >> 5)) * N + n0 + (tid & 31) * 4, 0, 1);

    // ---- A fragments: 16-bit A 16x32 layout. Lane half selects K-octet:
    //      V0..V3 = K off..off+7, V4..V7 = K 16+off..16+off+7  (off = laneHi*8)
    #pragma unroll
    for (int tm = 0; tm < 2; ++tm) {
      int m   = waveM * 32 + tm * 16 + laneLo;
      int off = laneHi * 8;
      a[tm].u[0] = *(const u32x4*)&As[p][m * KPAD + off];
      a[tm].u[1] = *(const u32x4*)&As[p][m * KPAD + 16 + off];
    }
    // ---- B fragments: 16-bit B 32x16 layout. Lane half selects K half:
    //      V0..V3 = K ks..ks+7, V4..V7 = K 16+ks..16+ks+7 (ks = laneHi*8); transposed Bs[n][k]
    #pragma unroll
    for (int tn = 0; tn < 4; ++tn) {
      int n  = waveN * 64 + tn * 16 + laneLo;
      int ks = laneHi * 8;
      b[tn].u[0] = *(const u32x4*)&Bs[p][n * KPAD + ks];
      b[tn].u[1] = *(const u32x4*)&Bs[p][n * KPAD + ks + 16];
    }

    #pragma unroll
    for (int tm = 0; tm < 2; ++tm)
      #pragma unroll
      for (int tn = 0; tn < 4; ++tn)
        acc[tm][tn] = __builtin_amdgcn_wmma_f32_16x16x32_bf16(
            false, a[tm].v, false, b[tn].v, (short)0, acc[tm][tn], false, false);

    // Convert + store next tile into the other buffer (after the WMMA block).
    if (kt + 1 < nk) commit(p ^ 1);

    __syncthreads();   // one barrier per K-step (protects both buffers)
  }

  // ---- epilogue: bias + optional ReLU, scatter by row map when routed ----
  // C/D layout: VGPR r -> M = r + laneHi*8, N = laneLo
  #pragma unroll
  for (int tm = 0; tm < 2; ++tm) {
    #pragma unroll
    for (int tn = 0; tn < 4; ++tn) {
      int n = n0 + waveN * 64 + tn * 16 + laneLo;
      float bv = bias[n];
      #pragma unroll
      for (int r = 0; r < 8; ++r) {
        int rl = waveM * 32 + tm * 16 + laneHi * 8 + r;
        int m  = GATHER ? rmap[rl] : (m0 + rl);
        if (m < 0) continue;
        float val = acc[tm][tn][r] + bv;
        if (RELU) val = fmaxf(val, 0.0f);
        C[(size_t)m * N + n] = val;
      }
    }
  }
}

// ---------------- Host-side launch ----------------
extern "C" void kernel_launch(void* const* d_in, const int* in_sizes, int n_in,
                              void* d_out, int out_size, void* d_ws, size_t ws_size,
                              hipStream_t stream) {
  (void)in_sizes; (void)n_in; (void)out_size; (void)ws_size;

  const float* x      = (const float*)d_in[0];
  const int*   labels = (const int*)  d_in[1];
  const float* W1  = (const float*)d_in[2];  const float* b1  = (const float*)d_in[3];
  const float* W2  = (const float*)d_in[4];  const float* b2  = (const float*)d_in[5];
  const float* EW1 = (const float*)d_in[6];  const float* Eb1 = (const float*)d_in[7];
  const float* EW2 = (const float*)d_in[8];  const float* Eb2 = (const float*)d_in[9];
  const float* DW1 = (const float*)d_in[10]; const float* Db1 = (const float*)d_in[11];
  const float* DW2 = (const float*)d_in[12]; const float* Db2 = (const float*)d_in[13];
  float* out = (float*)d_out;

  // Workspace layout (fp32): h1 (reused for decoder hidden), h2, he, sel, then ints.
  float* h1  = (float*)d_ws;                       // [8192,2048]  64 MB
  float* h2  = h1  + (size_t)B_SZ * DH;            // [8192, 512]  16 MB
  float* he  = h2  + (size_t)B_SZ * DBOT;          // [8192,1024]  32 MB
  float* sel = he  + (size_t)B_SZ * DEXP;          // [8192, 512]  16 MB
  int*   ib  = (int*)(sel + (size_t)B_SZ * DBOT);
  int* cnt  = ib;        // [16]
  int* off  = ib + 16;   // [16]
  int* cur  = ib + 32;   // [16]
  int* perm = ib + 48;   // [8192]

  // 1) Build class-grouped row permutation
  route_init   <<<1, 32, 0, stream>>>(cnt, cur);
  route_hist   <<<B_SZ / 256, 256, 0, stream>>>(labels, cnt, B_SZ);
  route_scan   <<<1, 32, 0, stream>>>(cnt, off);
  route_scatter<<<B_SZ / 256, 256, 0, stream>>>(labels, off, cur, perm, B_SZ);

  dim3 blk(256);

  // 2) Encoder
  gemm_bf16_wmma<true, false><<<dim3(B_SZ / BM, DH   / BN, 1), blk, 0, stream>>>(
      x,  W1, b1, h1, B_SZ, DH,   DIN, nullptr, nullptr, nullptr);
  gemm_bf16_wmma<true, false><<<dim3(B_SZ / BM, DBOT / BN, 1), blk, 0, stream>>>(
      h1, W2, b2, h2, B_SZ, DBOT, DH,  nullptr, nullptr, nullptr);

  // 3) Routed experts (only each sample's expert, not all 10)
  gemm_bf16_wmma<true, true><<<dim3(B_SZ / BM, DEXP / BN, NCLS), blk, 0, stream>>>(
      h2, EW1, Eb1, he,  B_SZ, DEXP, DBOT, perm, off, cnt);
  gemm_bf16_wmma<true, true><<<dim3(B_SZ / BM, DBOT / BN, NCLS), blk, 0, stream>>>(
      he, EW2, Eb2, sel, B_SZ, DBOT, DEXP, perm, off, cnt);

  // 4) Decoder (reuse h1 for decoder hidden)
  gemm_bf16_wmma<true,  false><<<dim3(B_SZ / BM, DH  / BN, 1), blk, 0, stream>>>(
      sel, DW1, Db1, h1,  B_SZ, DH,  DBOT, nullptr, nullptr, nullptr);
  gemm_bf16_wmma<false, false><<<dim3(B_SZ / BM, DIN / BN, 1), blk, 0, stream>>>(
      h1,  DW2, Db2, out, B_SZ, DIN, DH,   nullptr, nullptr, nullptr);
}
